// GraphAttentionLayer_81827716924079
// MI455X (gfx1250) — matrix operand order repaired and checked by reading, not compiled, not run on hardware
//
#include <hip/hip_runtime.h>
#include <hip/hip_bf16.h>

typedef __attribute__((ext_vector_type(16))) __bf16 v16bf;
typedef __attribute__((ext_vector_type(8)))  __bf16 v8bf;
typedef __attribute__((ext_vector_type(8)))  float  v8f;

#define NROWS 8192
#define DDIM  4096      // D_IN == D_OUT == 4096
#define ALPHA 0.2f

// Async copy of 16 bytes global -> LDS (gfx1250 GLOBAL_LOAD_ASYNC_TO_LDS_B128,
// tracked by ASYNCcnt). lds_off is the LDS byte offset, gptr the global addr.
__device__ __forceinline__ void async_copy_b128(unsigned lds_off, const void* gptr) {
    unsigned long long ga = (unsigned long long)gptr;
    asm volatile("global_load_async_to_lds_b128 %0, %1, off"
                 :: "v"(lds_off), "v"(ga) : "memory");
}

// ---------------------------------------------------------------------------
// Kernel 1: convert x and W from f32 to bf16 (grid-stride)
// ---------------------------------------------------------------------------
__global__ void cvt_kernel(const float* __restrict__ x, const float* __restrict__ W,
                           __bf16* __restrict__ xb, __bf16* __restrict__ wb) {
    const long nx = (long)NROWS * DDIM;
    const long nw = (long)DDIM * DDIM;
    long i = (long)blockIdx.x * blockDim.x + threadIdx.x;
    long stride = (long)gridDim.x * blockDim.x;
    for (long t = i; t < nx; t += stride) xb[t] = (__bf16)x[t];
    for (long t = i; t < nw; t += stride) wb[t] = (__bf16)W[t];
}

// ---------------------------------------------------------------------------
// Kernel 2: h = x @ W^T via bf16 WMMA, f32 accumulate.
// Workgroup tile 128(M) x 256(N); 8 waves arranged 2(M) x 4(N); wave tile
// 64x64 = 4x4 subtiles = 16 WMMAs per k-step (each frag reused 4x).
// Double-buffered LDS tiles (A 128x32 = 8 KB, B 256x32 = 16 KB) filled with
// global_load_async_to_lds_b128; s_wait_asynccnt overlaps next-tile copy with
// the current 16-WMMA block. Stores h transposed: hbT[o][i] = h[i][o].
// ---------------------------------------------------------------------------
__global__ __launch_bounds__(256)
void gemm1_kernel(const __bf16* __restrict__ xb, const __bf16* __restrict__ wb,
                  __bf16* __restrict__ hbT) {
    __shared__ __align__(64) __bf16 Abuf[2][128 * 32];   // 2 x 8 KB
    __shared__ __align__(64) __bf16 Bbuf[2][256 * 32];   // 2 x 16 KB

    const int t    = threadIdx.x;
    const int lane = t & 31;
    const int wave = t >> 5;             // 0..7
    const int wm   = wave >> 2;          // 0..1
    const int wn   = wave & 3;           // 0..3
    const int lm   = lane & 15;
    const int lh   = lane >> 4;          // 0 or 1
    const int rowg = blockIdx.y * 128;   // workgroup row base (x rows)
    const int colg = blockIdx.x * 256;   // workgroup col base (W rows)

    // Cooperative copy mapping: 4 consecutive lanes cover one tile row
    // (4 x 16 B = one 64 B row); instruction q advances 64 rows.
    const int crow = t >> 2;             // 0..63
    const int ckel = (t & 3) * 8;        // element offset 0,8,16,24
    const __bf16* gA = xb + (long)(rowg + crow) * DDIM + ckel;
    const __bf16* gB = wb + (long)(colg + crow) * DDIM + ckel;
    const unsigned lA0 = (unsigned)(unsigned long long)(const void*)&Abuf[0][crow * 32 + ckel];
    const unsigned lB0 = (unsigned)(unsigned long long)(const void*)&Bbuf[0][crow * 32 + ckel];

    auto issue_copy = [&](int buf, int kb) {
#pragma unroll
        for (int q = 0; q < 2; ++q)   // A: 128 rows
            async_copy_b128(lA0 + (unsigned)buf * 8192u + (unsigned)q * 4096u,
                            gA + kb + (long)q * 64 * DDIM);
#pragma unroll
        for (int q = 0; q < 4; ++q)   // B: 256 rows
            async_copy_b128(lB0 + (unsigned)buf * 16384u + (unsigned)q * 4096u,
                            gB + kb + (long)q * 64 * DDIM);
    };

    v8f acc[4][4] = {};
    issue_copy(0, 0);

    for (int kb = 0; kb < DDIM; kb += 32) {
        const int cur = (kb >> 5) & 1;
        if (kb + 32 < DDIM) {
            issue_copy(cur ^ 1, kb + 32);
            // 6 in-flight = the next buffer's copies; current buffer done.
            asm volatile("s_wait_asynccnt 0x6" ::: "memory");
        } else {
            asm volatile("s_wait_asynccnt 0x0" ::: "memory");
        }
        __syncthreads();

        v16bf a[4], b[4];
#pragma unroll
        for (int mt = 0; mt < 4; ++mt) {
            const __bf16* ap = &Abuf[cur][(wm * 64 + mt * 16 + lm) * 32 + lh * 8];
            v8bf alo = *(const v8bf*)(ap);
            v8bf ahi = *(const v8bf*)(ap + 16);
            a[mt] = __builtin_shufflevector(alo, ahi,
                     0,1,2,3,4,5,6,7,8,9,10,11,12,13,14,15);
        }
#pragma unroll
        for (int nt = 0; nt < 4; ++nt) {
            const __bf16* bp = &Bbuf[cur][(wn * 64 + nt * 16 + lm) * 32 + lh * 8];
            v8bf blo = *(const v8bf*)(bp);
            v8bf bhi = *(const v8bf*)(bp + 16);
            b[nt] = __builtin_shufflevector(blo, bhi,
                     0,1,2,3,4,5,6,7,8,9,10,11,12,13,14,15);
        }
#pragma unroll
        for (int mt = 0; mt < 4; ++mt)
#pragma unroll
            for (int nt = 0; nt < 4; ++nt)
                acc[mt][nt] = __builtin_amdgcn_wmma_f32_16x16x32_bf16(
                    false, a[mt], false, b[nt], (short)0, acc[mt][nt], false, false);

        __syncthreads();   // all reads of 'cur' done before it is refilled
    }

    // Epilogue: hbT[o][i] = bf16(acc). For fixed (mt,nt), r spans 8 consecutive
    // rows i -> pack into one v8bf contiguous store.
    const int row0 = rowg + wm * 64;
    const int col0 = colg + wn * 64;
#pragma unroll
    for (int mt = 0; mt < 4; ++mt) {
#pragma unroll
        for (int nt = 0; nt < 4; ++nt) {
            const int o = col0 + nt * 16 + lm;
            v8bf hv;
#pragma unroll
            for (int r = 0; r < 8; ++r) hv[r] = (__bf16)acc[mt][nt][r];
            *(v8bf*)(hbT + (long)o * NROWS + row0 + mt * 16 + lh * 8) = hv;
        }
    }
}

// ---------------------------------------------------------------------------
// Kernel 3: s1[i] = sum_o h[i][o]*a1[o], s2[i] = sum_o h[i][o]*a2[o]
// Coalesced reads over hbT (hbT[o][i] contiguous across threads in i).
// ---------------------------------------------------------------------------
__global__ __launch_bounds__(256)
void svec_kernel(const __bf16* __restrict__ hbT, const float* __restrict__ a,
                 float* __restrict__ s1, float* __restrict__ s2) {
    const int i = blockIdx.x * 256 + threadIdx.x;
    float acc1 = 0.f, acc2 = 0.f;
    for (int o = 0; o < DDIM; ++o) {
        float v = (float)hbT[(long)o * NROWS + i];
        acc1 += v * a[o];
        acc2 += v * a[DDIM + o];
    }
    s1[i] = acc1;
    s2[i] = acc2;
}

// ---------------------------------------------------------------------------
// Kernel 4: per-row online softmax stats over implicit e_ij = leaky(s1_i+s2_j).
// Thread i owns row i; s2 staged through LDS in 1024-element chunks.
// ---------------------------------------------------------------------------
__global__ __launch_bounds__(256)
void stats_kernel(const float* __restrict__ s1, const float* __restrict__ s2,
                  float* __restrict__ mrow, float* __restrict__ drow) {
    __shared__ float s2s[1024];
    const int i = blockIdx.x * 256 + threadIdx.x;
    const float s1v = s1[i];
    float m = -3.0e38f, dsum = 0.f;
    for (int jb = 0; jb < NROWS; jb += 1024) {
        __syncthreads();
        for (int t = threadIdx.x; t < 1024; t += 256) s2s[t] = s2[jb + t];
        __syncthreads();
        for (int jj = 0; jj < 1024; ++jj) {
            float e = s1v + s2s[jj];
            e = e > 0.f ? e : ALPHA * e;
            if (e > m) { dsum = dsum * __expf(m - e); m = e; }
            dsum += __expf(e - m);
        }
    }
    mrow[i] = m;
    drow[i] = dsum;
}

// ---------------------------------------------------------------------------
// Kernel 5: out = softmax(e) @ h, flash style. P tiles (64x32 bf16) are
// regenerated on the fly from s1/s2/m into LDS each K-step, then WMMA'd
// against hbT (B frags read contiguously from L2-resident hbT).
// Block = 8 waves sharing the same 64 rows; each wave owns 64 output cols
// (4x4 subtiles = 16 WMMAs per k-step).
// ---------------------------------------------------------------------------
__global__ __launch_bounds__(256)
void gemm2_kernel(const __bf16* __restrict__ hbT, const float* __restrict__ s1,
                  const float* __restrict__ s2, const float* __restrict__ mrow,
                  const float* __restrict__ drow, float* __restrict__ out) {
    __shared__ __align__(16) __bf16 Pbuf[64 * 32];
    const int lane = threadIdx.x & 31;
    const int wave = threadIdx.x >> 5;          // 0..7
    const int r0   = blockIdx.y * 64;
    const int c0   = blockIdx.x * 512 + wave * 64;
    const int lm   = lane & 15;
    const int lh   = lane >> 4;

    // P-generation mapping: 4 threads per row, 8 cols each
    const int prow = threadIdx.x >> 2;          // 0..63
    const int pcol = (threadIdx.x & 3) * 8;     // 0,8,16,24
    const float s1v = s1[r0 + prow];
    const float mv  = mrow[r0 + prow];

    v8f acc[4][4] = {};

    for (int jb = 0; jb < NROWS; jb += 32) {
        if (jb + 32 < NROWS)
            __builtin_prefetch(hbT + (long)(c0 + lm) * NROWS + jb + 32, 0, 1);
        __syncthreads();   // protect Pbuf readers of previous iteration
        v8bf pv;
#pragma unroll
        for (int q = 0; q < 8; ++q) {
            float e = s1v + s2[jb + pcol + q];
            e = e > 0.f ? e : ALPHA * e;
            pv[q] = (__bf16)__expf(e - mv);
        }
        *(v8bf*)(&Pbuf[prow * 32 + pcol]) = pv;
        __syncthreads();

        v16bf a[4], b[4];
#pragma unroll
        for (int mt = 0; mt < 4; ++mt) {
            const __bf16* ap = &Pbuf[(mt * 16 + lm) * 32 + lh * 8];
            v8bf alo = *(const v8bf*)(ap);
            v8bf ahi = *(const v8bf*)(ap + 16);
            a[mt] = __builtin_shufflevector(alo, ahi,
                     0,1,2,3,4,5,6,7,8,9,10,11,12,13,14,15);
        }
#pragma unroll
        for (int nt = 0; nt < 4; ++nt) {
            const __bf16* bp = hbT + (long)(c0 + nt * 16 + lm) * NROWS + jb + lh * 8;
            v8bf blo = *(const v8bf*)(bp);
            v8bf bhi = *(const v8bf*)(bp + 16);
            b[nt] = __builtin_shufflevector(blo, bhi,
                     0,1,2,3,4,5,6,7,8,9,10,11,12,13,14,15);
        }
#pragma unroll
        for (int mt = 0; mt < 4; ++mt)
#pragma unroll
            for (int nt = 0; nt < 4; ++nt)
                acc[mt][nt] = __builtin_amdgcn_wmma_f32_16x16x32_bf16(
                    false, a[mt], false, b[nt], (short)0, acc[mt][nt], false, false);
    }

    // Epilogue: scale by 1/d_i, f32 store
#pragma unroll
    for (int mt = 0; mt < 4; ++mt) {
#pragma unroll
        for (int r = 0; r < 8; ++r) {
            const int i = r0 + mt * 16 + r + lh * 8;
            const float inv = 1.0f / drow[i];
#pragma unroll
            for (int nt = 0; nt < 4; ++nt) {
                const int c = c0 + nt * 16 + lm;
                out[(long)i * DDIM + c] = acc[mt][nt][r] * inv;
            }
        }
    }
}

// ---------------------------------------------------------------------------
// Launch
// ---------------------------------------------------------------------------
extern "C" void kernel_launch(void* const* d_in, const int* in_sizes, int n_in,
                              void* d_out, int out_size, void* d_ws, size_t ws_size,
                              hipStream_t stream) {
    (void)in_sizes; (void)n_in; (void)out_size; (void)ws_size;
    const float* x = (const float*)d_in[0];   // (8192, 64, 64) f32
    const float* W = (const float*)d_in[1];   // (4096, 4096)  f32
    const float* a = (const float*)d_in[2];   // (1, 8192)     f32
    float* out = (float*)d_out;               // (8192, 4096)  f32

    char* ws = (char*)d_ws;
    __bf16* xb  = (__bf16*)(ws);                              // 64 MB
    __bf16* wb  = (__bf16*)(ws + (size_t)67108864);           // 32 MB
    __bf16* hbT = (__bf16*)(ws + (size_t)100663296);          // 64 MB, [4096][8192]
    float*  s1  = (float*)(ws + (size_t)167772160);           // 32 KB
    float*  s2  = s1 + NROWS;
    float*  mr  = s2 + NROWS;
    float*  dr  = mr + NROWS;

    cvt_kernel<<<2048, 256, 0, stream>>>(x, W, xb, wb);

    dim3 g1(DDIM / 256, NROWS / 128);   // (16, 64)
    gemm1_kernel<<<g1, 256, 0, stream>>>(xb, wb, hbT);

    svec_kernel<<<NROWS / 256, 256, 0, stream>>>(hbT, a, s1, s2);
    stats_kernel<<<NROWS / 256, 256, 0, stream>>>(s1, s2, mr, dr);

    dim3 g2(DDIM / 512, NROWS / 64);    // (8, 128)
    gemm2_kernel<<<g2, 256, 0, stream>>>(hbT, s1, s2, mr, dr, out);
}
